// LIIF_3839700763183
// MI455X (gfx1250) — matrix operand order repaired and checked by reading, not compiled, not run on hardware
//
#include <hip/hip_runtime.h>

typedef __attribute__((ext_vector_type(16))) _Float16 v16h;
typedef __attribute__((ext_vector_type(8)))  _Float16 v8h;
typedef __attribute__((ext_vector_type(8)))  float    v8f;

// Problem constants (match reference)
#define BQ      8192           // queries per image
#define CIMG    64
#define HIMG    128
#define K0      580            // 9*64 + 4
#define K0P     608            // padded to 19*32
#define KT0     19             // K-tiles for layer 0
#define KTH     4              // K-tiles for hidden layers (128/32)
#define NTH     8              // N-tiles for hidden layers (128/16)
#define NTO     4              // N-tiles for output layer (64/16)

// Workspace fragment offsets (in halfs).  Fragment = 32 lanes x 16 halfs = 512 halfs.
// L0: 19*8=152 frags, L1-3: 4*8=32 each, L4: 4*4=16.  Total 264 frags = 135168 halfs.
#define WOFF0 0
#define WOFF1 77824
#define WOFF2 94208
#define WOFF3 110592
#define WOFF4 126976

// ---------------------------------------------------------------------------
// Kernel 1: convert f32 weights -> f16 WMMA B-fragment layout in workspace.
// B-fragment (32x16, f16): lane l holds column n = l%16; element i holds
// K = (l/16)*8 + i (i<8) or 16 + (l/16)*8 + (i-8) (i>=8) within the 32-wide K tile.
// ---------------------------------------------------------------------------
__global__ __launch_bounds__(256)
void prep_weights(const float* __restrict__ w0, const float* __restrict__ w1,
                  const float* __restrict__ w2, const float* __restrict__ w3,
                  const float* __restrict__ w4, _Float16* __restrict__ ws) {
    int frag = blockIdx.x;            // 0..263
    const float* w; int ntiles, K, N, local; _Float16* base;
    if (frag < 152)      { w = w0; ntiles = 8; K = 580; N = 128; base = ws + WOFF0; local = frag;       }
    else if (frag < 184) { w = w1; ntiles = 8; K = 128; N = 128; base = ws + WOFF1; local = frag - 152; }
    else if (frag < 216) { w = w2; ntiles = 8; K = 128; N = 128; base = ws + WOFF2; local = frag - 184; }
    else if (frag < 248) { w = w3; ntiles = 8; K = 128; N = 128; base = ws + WOFF3; local = frag - 216; }
    else                 { w = w4; ntiles = 4; K = 128; N =  64; base = ws + WOFF4; local = frag - 248; }
    int kt = local / ntiles;
    int nt = local % ntiles;
    for (int e = threadIdx.x; e < 512; e += 256) {
        int l = e >> 4, i = e & 15;
        int n = nt * 16 + (l & 15);
        int k = kt * 32 + (i < 8 ? (l >> 4) * 8 + i : 16 + (l >> 4) * 8 + (i - 8));
        float v = (k < K) ? w[k * N + n] : 0.0f;
        base[(size_t)local * 512 + e] = (_Float16)v;
    }
}

// ---------------------------------------------------------------------------
// WMMA helpers
// A-fragment (16x32, f16) from LDS activations (row-major, `stride` halfs/row):
//   lane l: m = l%16, kh = l/16; elems 0..7  = K (kh*8 .. kh*8+7)
//                                elems 8..15 = K (16+kh*8 .. 16+kh*8+7)
// -> two 16-byte LDS loads per K-tile, software-pipelined one tile ahead so
// the ds_load latency hides under the previous v_wmma (avoids s_wait_dscnt 0).
// ---------------------------------------------------------------------------
__device__ __forceinline__
v8f wmma_mac(const _Float16* __restrict__ act, int stride, int ktiles,
             const _Float16* __restrict__ wf, int ntiles, int nt, int lane) {
    v8f acc = {};
    const int m  = lane & 15;
    const int kh = lane >> 4;
    const _Float16* arow = act + m * stride + kh * 8;
    const _Float16* wp   = wf + (size_t)nt * 512 + (size_t)lane * 16;

    // prologue: load K-tile 0
    v8h lo = *(const v8h*)(arow);
    v8h hi = *(const v8h*)(arow + 16);
#pragma unroll
    for (int kt = 0; kt < ktiles; ++kt) {
        // issue next tile's A loads before consuming the current one
        v8h lo_n = lo, hi_n = hi;
        if (kt + 1 < ktiles) {
            lo_n = *(const v8h*)(arow + (kt + 1) * 32);
            hi_n = *(const v8h*)(arow + (kt + 1) * 32 + 16);
        }
        v16h a;
#pragma unroll
        for (int i = 0; i < 8; ++i) { a[i] = lo[i]; a[i + 8] = hi[i]; }
        v16h bm = *(const v16h*)(wp + (size_t)kt * ntiles * 512);
        acc = __builtin_amdgcn_wmma_f32_16x16x32_f16(
            /*neg_a=*/false, a, /*neg_b=*/false, bm,
            /*c_mod=*/(short)0, acc, /*reuse_a=*/false, /*reuse_b=*/false);
        lo = lo_n; hi = hi_n;
    }
    return acc;
}

__device__ __forceinline__
void gemm_relu(const _Float16* __restrict__ src, int stride, int ktiles,
               const _Float16* __restrict__ wf, const float* __restrict__ bias,
               _Float16* __restrict__ dst, int lane, int wid) {
    v8f acc = wmma_mac(src, stride, ktiles, wf, NTH, wid, lane);
    const int n  = wid * 16 + (lane & 15);
    const int kh = lane >> 4;
    const float bv = bias[n];
#pragma unroll
    for (int r = 0; r < 8; ++r) {
        float v = acc[r] + bv;
        v = fmaxf(v, 0.0f);
        dst[(r + 8 * kh) * 128 + n] = (_Float16)v;
    }
}

// ---------------------------------------------------------------------------
// Main kernel: one block = 16 queries, 8 wave32s.  Loops over 4 corner shifts.
// ---------------------------------------------------------------------------
__global__ __launch_bounds__(256)
void liif_kernel(const float* __restrict__ x, const float* __restrict__ coord,
                 const float* __restrict__ cell,
                 const float* __restrict__ b0, const float* __restrict__ b1,
                 const float* __restrict__ b2, const float* __restrict__ b3,
                 const float* __restrict__ b4,
                 const _Float16* __restrict__ wfrag, float* __restrict__ out) {
    __shared__ _Float16 sBig[16 * K0P];    // layer-0 input, 16B-aligned rows (608*2=1216B)
    __shared__ _Float16 sBuf1[16 * 128];
    __shared__ _Float16 sBuf2[16 * 128];
    __shared__ float    sAcc[16 * 64];
    __shared__ float    sRel0[4][16], sRel1[4][16], sWgt[4][16], sAreaS[4][16];
    __shared__ float    sRC0[16], sRC1[16];
    __shared__ int      sIy[4][16], sIx[4][16];

    const int tid   = threadIdx.x;
    const int lane  = tid & 31;
    const int wid   = tid >> 5;
    const int bb    = blockIdx.x >> 9;           // batch index (512 blocks per image)
    const int qbase = (blockIdx.x & 511) << 4;   // 16 queries per block

    // ---- per-query, per-shift scalars (64 threads: s = t/16, q = t%16) ----
    if (tid < 64) {
        const int s = tid >> 4, q = tid & 15;
        const float vx = (s & 2) ? 1.0f : -1.0f;
        const float vy = (s & 1) ? 1.0f : -1.0f;
        const int gq = bb * BQ + qbase + q;
        const float c0r = coord[gq * 2 + 0];
        const float c1r = coord[gq * 2 + 1];
        const float rx = 1.0f / 128.0f;
        float c0 = fminf(fmaxf(c0r + vx * rx + 1e-6f, -1.0f + 1e-6f), 1.0f - 1e-6f);
        float c1 = fminf(fmaxf(c1r + vy * rx + 1e-6f, -1.0f + 1e-6f), 1.0f - 1e-6f);
        int iy = (int)fminf(fmaxf(rintf(((c0 + 1.0f) * 128.0f - 1.0f) * 0.5f), 0.0f), 127.0f);
        int ix = (int)fminf(fmaxf(rintf(((c1 + 1.0f) * 128.0f - 1.0f) * 0.5f), 0.0f), 127.0f);
        const float seqix = -1.0f + (2.0f * (float)ix + 1.0f) / 128.0f;
        const float seqiy = -1.0f + (2.0f * (float)iy + 1.0f) / 128.0f;
        const float r0 = (c0r - seqix) * 128.0f;
        const float r1 = (c1r - seqiy) * 128.0f;
        sIy[s][q] = iy; sIx[s][q] = ix;
        sRel0[s][q] = r0; sRel1[s][q] = r1;
        sAreaS[s][q] = fabsf(r0 * r1) + 1e-9f;
        if (s == 0) {
            sRC0[q] = cell[gq * 2 + 0] * 128.0f;
            sRC1[q] = cell[gq * 2 + 1] * 128.0f;
        }
    }
    for (int i = tid; i < 16 * 64; i += 256) sAcc[i] = 0.0f;
    __syncthreads();
    if (tid < 16) {
        const int q = tid;
        const float tot = sAreaS[0][q] + sAreaS[1][q] + sAreaS[2][q] + sAreaS[3][q];
#pragma unroll
        for (int s = 0; s < 4; ++s) sWgt[s][q] = sAreaS[3 - s][q] / tot;  // perm (3,2,1,0)
    }
    __syncthreads();

    // ------------------------- 4 corner shifts -------------------------
    for (int s = 0; s < 4; ++s) {
        // gather 16 x 576 f16 patch features (channel order c*9 + ky*3 + kx)
        for (int idx = tid; idx < 16 * 576; idx += 256) {
            const int q = idx / 576;
            const int f = idx - q * 576;
            const int c = f / 9;
            const int k = f - c * 9;
            const int ky = k / 3, kx = k - ky * 3;
            const int yy = sIy[s][q] + ky - 1;
            const int xx = sIx[s][q] + kx - 1;
            float v = 0.0f;
            if ((unsigned)yy < 128u && (unsigned)xx < 128u)
                v = x[(((size_t)bb * CIMG + c) * HIMG + yy) * HIMG + xx];
            sBig[q * K0P + f] = (_Float16)v;
        }
        if (tid < 16) {
            const int q = tid;
            sBig[q * K0P + 576] = (_Float16)sRel0[s][q];
            sBig[q * K0P + 577] = (_Float16)sRel1[s][q];
            sBig[q * K0P + 578] = (_Float16)sRC0[q];
            sBig[q * K0P + 579] = (_Float16)sRC1[q];
            for (int f = 580; f < K0P; ++f) sBig[q * K0P + f] = (_Float16)0.0f;
        }
        __syncthreads();

        // MLP: 580->128->128->128->128 (ReLU), each wave owns one 16-wide N tile
        gemm_relu(sBig,  K0P, KT0, wfrag + WOFF0, b0, sBuf1, lane, wid);
        __syncthreads();
        gemm_relu(sBuf1, 128, KTH, wfrag + WOFF1, b1, sBuf2, lane, wid);
        __syncthreads();
        gemm_relu(sBuf2, 128, KTH, wfrag + WOFF2, b2, sBuf1, lane, wid);
        __syncthreads();
        gemm_relu(sBuf1, 128, KTH, wfrag + WOFF3, b3, sBuf2, lane, wid);
        __syncthreads();

        // output layer 128->64, weighted accumulation (waves 0..3)
        if (wid < NTO) {
            v8f acc = wmma_mac(sBuf2, 128, KTH, wfrag + WOFF4, NTO, wid, lane);
            const int n  = wid * 16 + (lane & 15);
            const int kh = lane >> 4;
            const float bv = b4[n];
#pragma unroll
            for (int r = 0; r < 8; ++r) {
                const int M = r + 8 * kh;
                sAcc[M * 64 + n] += sWgt[s][M] * (acc[r] + bv);
            }
        }
        __syncthreads();
    }

    // ---- write [B, Q, 64] f32 output ----
    for (int i = tid; i < 16 * 64; i += 256) {
        const int q = i >> 6, c = i & 63;
        out[((size_t)bb * BQ + qbase + q) * 64 + c] = sAcc[i];
    }
}

// ---------------------------------------------------------------------------
extern "C" void kernel_launch(void* const* d_in, const int* in_sizes, int n_in,
                              void* d_out, int out_size, void* d_ws, size_t ws_size,
                              hipStream_t stream) {
    (void)in_sizes; (void)n_in; (void)out_size; (void)ws_size;
    const float* x     = (const float*)d_in[0];
    const float* coord = (const float*)d_in[1];
    const float* cell  = (const float*)d_in[2];
    const float* w0 = (const float*)d_in[3];  const float* b0 = (const float*)d_in[4];
    const float* w1 = (const float*)d_in[5];  const float* b1 = (const float*)d_in[6];
    const float* w2 = (const float*)d_in[7];  const float* b2 = (const float*)d_in[8];
    const float* w3 = (const float*)d_in[9];  const float* b3 = (const float*)d_in[10];
    const float* w4 = (const float*)d_in[11]; const float* b4 = (const float*)d_in[12];
    _Float16* wsH = (_Float16*)d_ws;          // needs 270,336 bytes
    float* out = (float*)d_out;

    prep_weights<<<264, 256, 0, stream>>>(w0, w1, w2, w3, w4, wsH);
    liif_kernel<<<1024, 256, 0, stream>>>(x, coord, cell, b0, b1, b2, b3, b4, wsH, out);
}
